// CrossTransformer_44873818309138
// MI455X (gfx1250) — matrix operand order repaired and checked by reading, not compiled, not run on hardware
//
#include <hip/hip_runtime.h>
#include <hip/hip_bf16.h>

// CrossTransformer for MI455X (gfx1250, wave32, WMMA).
// Heavy GEMMs (256->1024->256 MLP over 131072 neighbor positions) run on
// v_wmma_f32_16x16x32_bf16 with fp32 accumulation. One workgroup (8 waves)
// per query point; the 16 neighbors form the M=16 rows of every WMMA tile.

#define B_   4
#define C_   256
#define N_   2048
#define N2_  2048
#define M_   (N_ + N2_)
#define K_   16
#define PH_  64
#define AH_  1024
#define EPS_ 1e-5f
#define CNT_ ((float)(B_ * N_ * K_))   // BN statistic count

typedef __attribute__((ext_vector_type(16))) __bf16 v16bf;
typedef __attribute__((ext_vector_type(8)))  __bf16 v8bf;
typedef __attribute__((ext_vector_type(8)))  float  v8f;

static __device__ __forceinline__ v8f vzero8() {
  v8f z;
#pragma unroll
  for (int i = 0; i < 8; ++i) z[i] = 0.0f;
  return z;
}

// ---- WMMA fragment loaders (layouts per CDNA5 ISA 7.12.2) -----------------
// A (16x32 bf16): lane L: row m=L&15, half=L>>4; elems 0..7 <-> K=half*8..+7,
// elems 8..15 <-> K=16+half*8..+7.  Source: LDS row-major [16][KD].
static __device__ __forceinline__ v16bf ldsA(const __bf16* Xs, int KD, int kk, int lane) {
  const int m = lane & 15, half = lane >> 4;
  const __bf16* p = Xs + m * KD + kk * 32 + half * 8;
  v8bf r0 = *reinterpret_cast<const v8bf*>(p);
  v8bf r1 = *reinterpret_cast<const v8bf*>(p + 16);
  return __builtin_shufflevector(r0, r1, 0,1,2,3,4,5,6,7,8,9,10,11,12,13,14,15);
}

// B (32x16 bf16): lane L: col n=L&15, half=L>>4; elems 0..7 <-> K=half*8..+7,
// elems 8..15 <-> K=16+half*8..+7.  Source: global weight, row-major [OUT][KD]; column n maps
// to output row outBase+n, so each lane reads two 8-element bf16 runs.
static __device__ __forceinline__ v16bf ldgB(const __bf16* W, int KD, int outBase,
                                             int kk, int lane) {
  const int n = lane & 15, half = lane >> 4;
  const __bf16* p = W + (size_t)(outBase + n) * KD + kk * 32 + half * 8;
  v8bf r0 = *reinterpret_cast<const v8bf*>(p);
  v8bf r1 = *reinterpret_cast<const v8bf*>(p + 16);
  return __builtin_shufflevector(r0, r1, 0,1,2,3,4,5,6,7,8,9,10,11,12,13,14,15);
}

static __device__ __forceinline__ v8f wmma_bf16(v16bf a, v16bf b, v8f c) {
  return __builtin_amdgcn_wmma_f32_16x16x32_bf16(false, a, false, b, (short)0, c,
                                                 false, false);
}

// ---- k_prep: weights -> bf16, zero stat accumulators ----------------------
__global__ void k_prep(const float* pw2, const float* aw1, const float* aw2,
                       __bf16* pw2b, __bf16* aw1b, __bf16* aw2b,
                       float* pstats, float* astats) {
  const int i = blockIdx.x * 256 + threadIdx.x;
  if (i < C_ * PH_) pw2b[i] = (__bf16)pw2[i];
  if (i < AH_ * C_) aw1b[i] = (__bf16)aw1[i];
  if (i < C_ * AH_) aw2b[i] = (__bf16)aw2[i];
  if (i < 2 * PH_) pstats[i] = 0.0f;   // [0:64) sum, [64:128) sumsq
  if (i < 2 * AH_) astats[i] = 0.0f;   // [0:1024) sum, [1024:2048) sumsq
}

// ---- k_transpose: fusion_feat_t[b][m][c] (contiguous channel vectors) -----
__global__ void k_transpose(const float* feat, const float* featdb, float* fft) {
  const int bm = blockIdx.x;
  const int b = bm / M_, m = bm % M_;
  const int c = threadIdx.x;
  const float v = (m < N_) ? feat[((size_t)b * C_ + c) * N_ + m]
                           : featdb[((size_t)b * C_ + c) * N2_ + (m - N_)];
  fft[((size_t)b * M_ + m) * C_ + c] = v;
}

// ---- k_knn: per-thread register top-16 over 4096 LDS-staged points --------
__global__ void k_knn(const float* pcd, const float* pcddb, int* idx) {
  __shared__ float px[M_], py[M_], pz[M_];   // 48 KB
  const int tid = threadIdx.x;
  const int b  = blockIdx.x / (N_ / 256);
  const int n  = (blockIdx.x % (N_ / 256)) * 256 + tid;

  for (int i = tid; i < M_; i += 256) {
    if (i < N_) {
      px[i] = pcd[((size_t)b * 3 + 0) * N_ + i];
      py[i] = pcd[((size_t)b * 3 + 1) * N_ + i];
      pz[i] = pcd[((size_t)b * 3 + 2) * N_ + i];
    } else {
      const int j = i - N_;
      px[i] = pcddb[((size_t)b * 3 + 0) * N2_ + j];
      py[i] = pcddb[((size_t)b * 3 + 1) * N2_ + j];
      pz[i] = pcddb[((size_t)b * 3 + 2) * N2_ + j];
    }
  }
  __syncthreads();

  const float qx = pcd[((size_t)b * 3 + 0) * N_ + n];
  const float qy = pcd[((size_t)b * 3 + 1) * N_ + n];
  const float qz = pcd[((size_t)b * 3 + 2) * N_ + n];

  float bd[K_]; int bi[K_];
#pragma unroll
  for (int j = 0; j < K_; ++j) { bd[j] = 3.4e38f; bi[j] = 0; }

  for (int m = 0; m < M_; ++m) {
    const float dx = qx - px[m], dy = qy - py[m], dz = qz - pz[m];
    float d = dx * dx + dy * dy + dz * dz;
    if (d < bd[K_ - 1]) {               // sorted swap-chain insert (registers)
      int id = m;
#pragma unroll
      for (int j = 0; j < K_; ++j) {
        if (d < bd[j]) {
          float td = bd[j]; int ti = bi[j];
          bd[j] = d; bi[j] = id;
          d = td; id = ti;
        }
      }
    }
  }
  const size_t base = ((size_t)b * N_ + n) * K_;
#pragma unroll
  for (int j = 0; j < K_; ++j) idx[base + j] = bi[j];
}

// ---- k_pos_hidden: conv 3->64 (+bias), store fp32, accumulate BN stats ----
__global__ void k_pos_hidden(const float* pcd, const float* pcddb, const int* idx,
                             const float* pw1, const float* pb1,
                             float* hid1, float* pstats) {
  __shared__ float ls[PH_], lss[PH_];
  const int tid = threadIdx.x;
  if (tid < PH_) { ls[tid] = 0.0f; lss[tid] = 0.0f; }
  __syncthreads();

  const int pos = blockIdx.x * 256 + tid;
  const int b = pos / (N_ * K_);
  const int rem = pos % (N_ * K_);
  const int n = rem / K_;
  const int m = idx[pos];

  const float qx = pcd[((size_t)b * 3 + 0) * N_ + n];
  const float qy = pcd[((size_t)b * 3 + 1) * N_ + n];
  const float qz = pcd[((size_t)b * 3 + 2) * N_ + n];
  float gx, gy, gz;
  if (m < N_) {
    gx = pcd[((size_t)b * 3 + 0) * N_ + m];
    gy = pcd[((size_t)b * 3 + 1) * N_ + m];
    gz = pcd[((size_t)b * 3 + 2) * N_ + m];
  } else {
    const int j = m - N_;
    gx = pcddb[((size_t)b * 3 + 0) * N2_ + j];
    gy = pcddb[((size_t)b * 3 + 1) * N2_ + j];
    gz = pcddb[((size_t)b * 3 + 2) * N2_ + j];
  }
  const float dx = qx - gx, dy = qy - gy, dz = qz - gz;

  const size_t obase = (size_t)pos * PH_;
#pragma unroll 8
  for (int h = 0; h < PH_; ++h) {
    const float hv = pw1[h * 3 + 0] * dx + pw1[h * 3 + 1] * dy +
                     pw1[h * 3 + 2] * dz + pb1[h];
    hid1[obase + h] = hv;
    atomicAdd(&ls[h], hv);
    atomicAdd(&lss[h], hv * hv);
  }
  __syncthreads();
  if (tid < PH_) {
    atomicAdd(&pstats[tid], ls[tid]);
    atomicAdd(&pstats[PH_ + tid], lss[tid]);
  }
}

__global__ void k_fin_pos(float* pstats, const float* g, const float* be) {
  const int t = threadIdx.x;           // 64 threads
  const float mean = pstats[t] / CNT_;
  const float var  = pstats[PH_ + t] / CNT_ - mean * mean;
  const float sc   = g[t] * rsqrtf(var + EPS_);
  pstats[2 * PH_ + t] = sc;                       // scale
  pstats[3 * PH_ + t] = be[t] - mean * sc;        // shift
}

// ---- k_pos_emb: BN+relu then WMMA 64->256; emit attn_in & value (bf16) ----
__global__ void k_pos_emb(const float* hid1, const int* idx, const float* fft,
                          const __bf16* pw2b, const float* pb2, const float* pstats,
                          __bf16* attn_in, __bf16* val) {
  __shared__ __align__(16) __bf16 Hp[K_ * PH_];   // 2 KB
  __shared__ float pe[K_ * C_];                   // 16 KB
  __shared__ int lidx[K_];

  const int bn = blockIdx.x;
  const int b = bn / N_, n = bn % N_;
  const int tid = threadIdx.x;
  const int wid = tid >> 5, lane = tid & 31;
  const int n16 = lane & 15, half = lane >> 4;

  const float* psc = pstats + 2 * PH_;
  const float* psh = pstats + 3 * PH_;
  const size_t hbase = (size_t)bn * K_ * PH_;
  for (int i = tid; i < K_ * PH_; i += 256) {
    const int h = i % PH_;
    const float y = hid1[hbase + i] * psc[h] + psh[h];
    Hp[i] = (__bf16)fmaxf(y, 0.0f);
  }
  if (tid < K_) lidx[tid] = idx[(size_t)bn * K_ + tid];
  __syncthreads();

  v8f acc[2] = {vzero8(), vzero8()};
#pragma unroll
  for (int kk = 0; kk < PH_ / 32; ++kk) {
    const v16bf a = ldsA(Hp, PH_, kk, lane);
#pragma unroll
    for (int t = 0; t < 2; ++t) {
      const v16bf bm = ldgB(pw2b, PH_, (wid * 2 + t) * 16, kk, lane);
      acc[t] = wmma_bf16(a, bm, acc[t]);
    }
  }
#pragma unroll
  for (int t = 0; t < 2; ++t) {
    const int c = (wid * 2 + t) * 16 + n16;
    const float bias = pb2[c];
#pragma unroll
    for (int v = 0; v < 8; ++v) {
      const int mr = v + 8 * half;               // D: VGPR v, half -> row M
      pe[mr * C_ + c] = acc[t][v] + bias;
    }
  }
  __syncthreads();

  const int c = tid;
  const float q = fft[((size_t)b * M_ + n) * C_ + c];
  const size_t obase = (size_t)bn * K_ * C_;
#pragma unroll
  for (int k = 0; k < K_; ++k) {
    const float g = fft[((size_t)b * M_ + lidx[k]) * C_ + c];
    const float p = pe[k * C_ + c];
    attn_in[obase + k * C_ + c] = (__bf16)(q - g + p);
    val[obase + k * C_ + c]     = (__bf16)(g + p);
  }
}

// ---- GEMM1: H(16x1024) = X(16x256) * W1^T, 64 WMMAs per wave --------------
static __device__ __forceinline__ void gemm1(const __bf16* Xs, const __bf16* aw1b,
                                             int wid, int lane, v8f acc[8]) {
#pragma unroll
  for (int t = 0; t < 8; ++t) acc[t] = vzero8();
#pragma unroll
  for (int kk = 0; kk < C_ / 32; ++kk) {
    const v16bf a = ldsA(Xs, C_, kk, lane);
#pragma unroll
    for (int t = 0; t < 8; ++t) {
      const v16bf bm = ldgB(aw1b, C_, (wid * 8 + t) * 16, kk, lane);
      acc[t] = wmma_bf16(a, bm, acc[t]);
    }
  }
}

// ---- k_attn_stats: GEMM1 once, reduce rows, atomic channel sums -----------
__global__ void k_attn_stats(const __bf16* attn_in, const __bf16* aw1b,
                             const float* ab1, float* astats) {
  __shared__ __align__(16) __bf16 Xs[K_ * C_];    // 8 KB
  const int tid = threadIdx.x;
  const int wid = tid >> 5, lane = tid & 31;
  const size_t ibase = (size_t)blockIdx.x * K_ * C_;

  for (int i = tid; i < (K_ * C_) / 8; i += 256)
    reinterpret_cast<v8bf*>(Xs)[i] = reinterpret_cast<const v8bf*>(attn_in + ibase)[i];
  __syncthreads();

  v8f acc[8];
  gemm1(Xs, aw1b, wid, lane, acc);

  const int n16 = lane & 15;
#pragma unroll
  for (int t = 0; t < 8; ++t) {
    const int c = (wid * 8 + t) * 16 + n16;
    const float bias = ab1[c];
    float s = 0.0f, s2 = 0.0f;
#pragma unroll
    for (int v = 0; v < 8; ++v) {
      const float x = acc[t][v] + bias;
      s += x; s2 += x * x;
    }
    s  += __shfl_xor(s, 16, 32);    // partner lane holds the other 8 rows
    s2 += __shfl_xor(s2, 16, 32);
    if (lane < 16) {
      atomicAdd(&astats[c], s);
      atomicAdd(&astats[AH_ + c], s2);
    }
  }
}

__global__ void k_fin_attn(float* astats, const float* g, const float* be) {
  const int c = blockIdx.x * 256 + threadIdx.x;
  const float mean = astats[c] / CNT_;
  const float var  = astats[AH_ + c] / CNT_ - mean * mean;
  const float sc   = g[c] * rsqrtf(var + EPS_);
  astats[2 * AH_ + c] = sc;
  astats[3 * AH_ + c] = be[c] - mean * sc;
}

// ---- k_attn_fused: GEMM1 -> BN/relu -> GEMM2 -> softmax(k) -> output ------
__global__ void k_attn_fused(const __bf16* attn_in, const __bf16* val,
                             const __bf16* aw1b, const __bf16* aw2b,
                             const float* ab1, const float* ab2,
                             const float* astats, float* out) {
  // 56 KB carve: Ls (16 KB, overlaps dead Xs) | Vs (8 KB) | Hs (32 KB)
  __shared__ __align__(16) char smem[16384 + 8192 + 32768];
  float*  Ls = reinterpret_cast<float*>(smem);
  __bf16* Xs = reinterpret_cast<__bf16*>(smem);
  __bf16* Vs = reinterpret_cast<__bf16*>(smem + 16384);
  __bf16* Hs = reinterpret_cast<__bf16*>(smem + 16384 + 8192);

  const int bn = blockIdx.x;
  const int b = bn / N_, n = bn % N_;
  const int tid = threadIdx.x;
  const int wid = tid >> 5, lane = tid & 31;
  const int n16 = lane & 15, half = lane >> 4;
  const size_t ibase = (size_t)bn * K_ * C_;

  for (int i = tid; i < (K_ * C_) / 8; i += 256) {
    reinterpret_cast<v8bf*>(Xs)[i] = reinterpret_cast<const v8bf*>(attn_in + ibase)[i];
    reinterpret_cast<v8bf*>(Vs)[i] = reinterpret_cast<const v8bf*>(val + ibase)[i];
  }
  __syncthreads();

  v8f acc[8];
  gemm1(Xs, aw1b, wid, lane, acc);

  // BN (bias folded) + relu -> Hs[16][1024] bf16
#pragma unroll
  for (int t = 0; t < 8; ++t) {
    const int c = (wid * 8 + t) * 16 + n16;
    const float sc = astats[2 * AH_ + c];
    const float sh = ab1[c] * sc + astats[3 * AH_ + c];
#pragma unroll
    for (int v = 0; v < 8; ++v) {
      const int mr = v + 8 * half;
      Hs[mr * AH_ + c] = (__bf16)fmaxf(acc[t][v] * sc + sh, 0.0f);
    }
  }
  __syncthreads();   // all waves done with Xs + Hs complete

  // GEMM2: L(16x256) = Hrelu(16x1024) * W2^T, 64 WMMAs per wave
  v8f acc2[2] = {vzero8(), vzero8()};
#pragma unroll 4
  for (int kk = 0; kk < AH_ / 32; ++kk) {
    const v16bf a = ldsA(Hs, AH_, kk, lane);
#pragma unroll
    for (int t = 0; t < 2; ++t) {
      const v16bf bm = ldgB(aw2b, AH_, (wid * 2 + t) * 16, kk, lane);
      acc2[t] = wmma_bf16(a, bm, acc2[t]);
    }
  }
#pragma unroll
  for (int t = 0; t < 2; ++t) {
    const int c = (wid * 2 + t) * 16 + n16;
    const float bias = ab2[c];
#pragma unroll
    for (int v = 0; v < 8; ++v) {
      const int mr = v + 8 * half;
      Ls[mr * C_ + c] = acc2[t][v] + bias;
    }
  }
  __syncthreads();

  // softmax over k per channel, then weighted sum with value
  const int c = tid;
  float mx = -3.4e38f;
#pragma unroll
  for (int k = 0; k < K_; ++k) mx = fmaxf(mx, Ls[k * C_ + c]);
  float e[K_], s = 0.0f;
#pragma unroll
  for (int k = 0; k < K_; ++k) { e[k] = __expf(Ls[k * C_ + c] - mx); s += e[k]; }
  const float inv = 1.0f / s;
  float o = 0.0f;
#pragma unroll
  for (int k = 0; k < K_; ++k) o += e[k] * (float)Vs[k * C_ + c];
  out[((size_t)b * C_ + c) * N_ + n] = o * inv;
}

// ---------------------------------------------------------------------------
extern "C" void kernel_launch(void* const* d_in, const int* in_sizes, int n_in,
                              void* d_out, int out_size, void* d_ws, size_t ws_size,
                              hipStream_t stream) {
  const float* pcd      = (const float*)d_in[0];
  const float* feat     = (const float*)d_in[1];
  const float* pcddb    = (const float*)d_in[2];
  const float* featdb   = (const float*)d_in[3];
  const float* pos_w1   = (const float*)d_in[4];
  const float* pos_b1   = (const float*)d_in[5];
  const float* pos_g1   = (const float*)d_in[6];
  const float* pos_be1  = (const float*)d_in[7];
  const float* pos_w2   = (const float*)d_in[8];
  const float* pos_b2   = (const float*)d_in[9];
  const float* attn_w1  = (const float*)d_in[10];
  const float* attn_b1  = (const float*)d_in[11];
  const float* attn_g1  = (const float*)d_in[12];
  const float* attn_be1 = (const float*)d_in[13];
  const float* attn_w2  = (const float*)d_in[14];
  const float* attn_b2  = (const float*)d_in[15];
  float* out = (float*)d_out;

  char* ws = (char*)d_ws;
  size_t off = 0;
  auto take = [&](size_t bytes) -> void* {
    off = (off + 255) & ~(size_t)255;
    void* p = ws + off;
    off += bytes;
    return p;
  };
  const size_t BNK = (size_t)B_ * N_ * K_;
  int*    idxp    = (int*)   take(BNK * sizeof(int));
  float*  fft     = (float*) take((size_t)B_ * M_ * C_ * sizeof(float));     // 16.8 MB
  float*  hid1    = (float*) take(BNK * PH_ * sizeof(float));                // 33.6 MB
  __bf16* attn_in = (__bf16*)take(BNK * C_ * sizeof(__bf16));                // 67 MB
  __bf16* valb    = (__bf16*)take(BNK * C_ * sizeof(__bf16));                // 67 MB
  __bf16* pw2b    = (__bf16*)take((size_t)C_ * PH_ * sizeof(__bf16));
  __bf16* aw1b    = (__bf16*)take((size_t)AH_ * C_ * sizeof(__bf16));
  __bf16* aw2b    = (__bf16*)take((size_t)C_ * AH_ * sizeof(__bf16));
  float*  pstats  = (float*) take(4 * PH_ * sizeof(float));
  float*  astats  = (float*) take(4 * AH_ * sizeof(float));
  (void)ws_size; (void)in_sizes; (void)n_in; (void)out_size;  // ~186 MB used

  k_prep      <<<(AH_ * C_) / 256, 256, 0, stream>>>(pos_w2, attn_w1, attn_w2,
                                                     pw2b, aw1b, aw2b, pstats, astats);
  k_transpose <<<B_ * M_,          256, 0, stream>>>(feat, featdb, fft);
  k_knn       <<<B_ * (N_ / 256),  256, 0, stream>>>(pcd, pcddb, idxp);
  k_pos_hidden<<<(int)(BNK / 256), 256, 0, stream>>>(pcd, pcddb, idxp, pos_w1, pos_b1,
                                                     hid1, pstats);
  k_fin_pos   <<<1,                 64, 0, stream>>>(pstats, pos_g1, pos_be1);
  k_pos_emb   <<<B_ * N_,          256, 0, stream>>>(hid1, idxp, fft, pw2b, pos_b2,
                                                     pstats, attn_in, valb);
  k_attn_stats<<<B_ * N_,          256, 0, stream>>>(attn_in, aw1b, attn_b1, astats);
  k_fin_attn  <<<AH_ / 256,        256, 0, stream>>>(astats, attn_g1, attn_be1);
  k_attn_fused<<<B_ * N_,          256, 0, stream>>>(attn_in, valb, aw1b, aw2b,
                                                     attn_b1, attn_b2, astats, out);
}
